// QuantizedLinear_21182778703901
// MI455X (gfx1250) — compile-verified, matrix operand order checked by reading, not run on hardware
//
#include <hip/hip_runtime.h>
#include <stdint.h>

// Problem constants (match reference)
#define IN_FEATURES   4096
#define OUT_FEATURES  4096
#define NROWS         4096   // 2 * 2048 batch rows
#define GROUP_SIZE    128

// GEMM tiling
#define BM 128
#define BN 128
#define BK 32
#define LDK (BK + 8)         // padded LDS row stride in bf16 elements (80 B, 16B-aligned)

typedef __attribute__((ext_vector_type(16))) __bf16 v16bf;
typedef __attribute__((ext_vector_type(8)))  __bf16 v8bf;
typedef __attribute__((ext_vector_type(8)))  float  v8f;

// ---------------------------------------------------------------------------
// CDNA5 async copy: global -> LDS, 16 B per lane, tracked by ASYNCcnt.
// (CDNA5 ISA; VGLOBAL opcode 98)
// ---------------------------------------------------------------------------
__device__ __forceinline__ void async_copy_b128(void* lds_dst, const void* gsrc) {
  asm volatile("global_load_async_to_lds_b128 %0, %1, off"
               :
               : "v"((uint32_t)(uintptr_t)lds_dst), "v"(gsrc)
               : "memory");
}
__device__ __forceinline__ void wait_async0() {
  asm volatile("s_wait_asynccnt 0x0" ::: "memory");
}

// Load one 16x32 bf16 A/B fragment from padded LDS.
// ISA 7.12.2 16-bit layout: lanes 0-15 -> rows, K in {0..7,16..23};
// lanes 16-31 -> same rows, K in {8..15,24..31}.
__device__ __forceinline__ v16bf load_frag(const __bf16* lds, int rowBase, int lane) {
  const int r  = rowBase + (lane & 15);
  const int kb = (lane >> 4) * 8;
  const __bf16* p = lds + r * LDK;
  v16bf f;
#pragma unroll
  for (int e = 0; e < 8; ++e) f[e] = p[kb + e];
#pragma unroll
  for (int e = 0; e < 8; ++e) f[8 + e] = p[16 + kb + e];
  return f;
}

// ===========================================================================
// Prepass A: fp32 -> bf16 straight conversion (x). 16 elems/thread.
// ===========================================================================
__global__ __launch_bounds__(256)
void cvt_f32_bf16(const float* __restrict__ src, __bf16* __restrict__ dst) {
  const size_t base = ((size_t)blockIdx.x * 256 + threadIdx.x) * 16;
#pragma unroll
  for (int i = 0; i < 2; ++i) {
    const float4 a = *(const float4*)(src + base + i * 8);
    const float4 b = *(const float4*)(src + base + i * 8 + 4);
    v8bf o;
    o[0] = (__bf16)a.x; o[1] = (__bf16)a.y; o[2] = (__bf16)a.z; o[3] = (__bf16)a.w;
    o[4] = (__bf16)b.x; o[5] = (__bf16)b.y; o[6] = (__bf16)b.z; o[7] = (__bf16)b.w;
    *(v8bf*)(dst + base + i * 8) = o;
  }
}

// ===========================================================================
// Prepass B: fp32 H[k][n] -> bf16 Ht[n][k] (tiled transpose via LDS)
// ===========================================================================
#define TT 64
__global__ __launch_bounds__(256)
void transpose_cvt_f32_bf16(const float* __restrict__ src,  // [K][O]
                            __bf16* __restrict__ dst) {     // [O][K]
  __shared__ __align__(16) __bf16 tile[TT][TT + 8];          // row stride 144 B
  const int t = threadIdx.x;
  const int kBase = blockIdx.y * TT;
  const int nBase = blockIdx.x * TT;
#pragma unroll
  for (int i = 0; i < 4; ++i) {
    const int idx = t + i * 256;        // 1024 float4 chunks
    const int r   = idx >> 4;           // 0..63 (k)
    const int c4  = idx & 15;           // 0..15 (n/4)
    const float4 v = *(const float4*)(src + (size_t)(kBase + r) * OUT_FEATURES + nBase + c4 * 4);
    const int c = c4 * 4;
    tile[c + 0][r] = (__bf16)v.x;
    tile[c + 1][r] = (__bf16)v.y;
    tile[c + 2][r] = (__bf16)v.z;
    tile[c + 3][r] = (__bf16)v.w;
  }
  __syncthreads();
#pragma unroll
  for (int i = 0; i < 2; ++i) {
    const int idx = t + i * 256;        // 512 v8bf chunks
    const int n   = idx >> 3;           // 0..63
    const int c   = idx & 7;            // 0..7
    const v8bf o = *(const v8bf*)&tile[n][c * 8];
    *(v8bf*)(dst + (size_t)(nBase + n) * IN_FEATURES + kBase + c * 8) = o;
  }
}

// ===========================================================================
// GEMM1 (fast path): xr[m][n] = sum_k xb[m][k] * ht[n][k], all-bf16 operands,
// both tiles via async global->LDS copies, double-buffered.
// ===========================================================================
__global__ __launch_bounds__(256)
void gemm_bf16_async(const __bf16* __restrict__ A,   // [M][K]
                     const __bf16* __restrict__ Bt,  // [N][K]
                     __bf16* __restrict__ C) {       // [M][N] bf16
  __shared__ __align__(16) __bf16 sa[2][BM * LDK];
  __shared__ __align__(16) __bf16 sb[2][BN * LDK];

  const int t     = threadIdx.x;
  const int lane  = t & 31;
  const int wid   = t >> 5;
  const int waveM = (wid & 3) * 32;   // 4 waves along M
  const int waveN = (wid >> 2) * 64;  // 2 waves along N
  const int rowBase = blockIdx.y * BM;
  const int colBase = blockIdx.x * BN;

  auto issue = [&](int buf, int k0) {
#pragma unroll
    for (int i = 0; i < 2; ++i) {
      const int idx = t + i * 256;    // 512 chunks of 16B per operand tile
      const int row = idx >> 2;       // 0..127
      const int c   = idx & 3;        // 0..3 (k/8)
      async_copy_b128(&sa[buf][row * LDK + c * 8],
                      A + (size_t)(rowBase + row) * IN_FEATURES + k0 + c * 8);
      async_copy_b128(&sb[buf][row * LDK + c * 8],
                      Bt + (size_t)(colBase + row) * IN_FEATURES + k0 + c * 8);
    }
  };

  v8f acc[2][4] = {};
  issue(0, 0);

  const int NKB = IN_FEATURES / BK;
  for (int kb = 0; kb < NKB; ++kb) {
    const int cur = kb & 1;
    wait_async0();
    __syncthreads();
    if (kb + 1 < NKB) issue(cur ^ 1, (kb + 1) * BK);

    v16bf af[2], bfr[4];
#pragma unroll
    for (int i = 0; i < 2; ++i) af[i]  = load_frag(sa[cur], waveM + i * 16, lane);
#pragma unroll
    for (int j = 0; j < 4; ++j) bfr[j] = load_frag(sb[cur], waveN + j * 16, lane);
#pragma unroll
    for (int i = 0; i < 2; ++i)
#pragma unroll
      for (int j = 0; j < 4; ++j)
        acc[i][j] = __builtin_amdgcn_wmma_f32_16x16x32_bf16(
            false, af[i], false, bfr[j], (short)0, acc[i][j], false, false);
  }

  const int ln = lane & 15;
  const int hi = lane >> 4;
#pragma unroll
  for (int i = 0; i < 2; ++i)
#pragma unroll
    for (int j = 0; j < 4; ++j)
#pragma unroll
      for (int g = 0; g < 8; ++g) {
        const int m = rowBase + waveM + i * 16 + hi * 8 + g;
        const int n = colBase + waveN + j * 16 + ln;
        C[(size_t)m * OUT_FEATURES + n] = (__bf16)acc[i][j][g];
      }
}

// ===========================================================================
// GEMM2 (fast path): out = xr @ Wt + bias. A tile async-copied; W tile
// dequantized int4 -> bf16 through VGPRs into LDS. Double-buffered.
// ===========================================================================
__global__ __launch_bounds__(256)
void qlinear_gemm_async(const __bf16* __restrict__ xr,
                        const int*    __restrict__ wq,
                        const float*  __restrict__ scales,
                        const float*  __restrict__ bias,
                        float* __restrict__ out) {
  __shared__ __align__(16) __bf16 sa[2][BM * LDK];
  __shared__ __align__(16) __bf16 sw[2][BN * LDK];

  const int t     = threadIdx.x;
  const int lane  = t & 31;
  const int wid   = t >> 5;
  const int waveM = (wid & 3) * 32;
  const int waveN = (wid >> 2) * 64;
  const int rowBase = blockIdx.y * BM;
  const int colBase = blockIdx.x * BN;

  auto issueA = [&](int buf, int k0) {
#pragma unroll
    for (int i = 0; i < 2; ++i) {
      const int idx = t + i * 256;
      const int row = idx >> 2;
      const int c   = idx & 3;
      async_copy_b128(&sa[buf][row * LDK + c * 8],
                      xr + (size_t)(rowBase + row) * IN_FEATURES + k0 + c * 8);
    }
  };
  auto loadW = [&](int buf, int k0) {
#pragma unroll
    for (int i = 0; i < 2; ++i) {
      const int idx = t + i * 256;    // 512 int4 chunks
      const int o   = idx >> 2;       // 0..127
      const int c   = idx & 3;
      const int og  = colBase + o;
      const int4 b4 = *(const int4*)(wq + (size_t)og * (IN_FEATURES / 2) + (k0 >> 1) + c * 4);
      const float s = scales[og * (IN_FEATURES / GROUP_SIZE) + (k0 >> 7)];
      __bf16* d = &sw[buf][o * LDK + c * 8];
      const int bb[4] = { b4.x, b4.y, b4.z, b4.w };
#pragma unroll
      for (int q = 0; q < 4; ++q) {
        const int b = bb[q];
        d[2 * q + 0] = (__bf16)((float)(( b        & 15) - 8) * s);
        d[2 * q + 1] = (__bf16)((float)(((b >> 4)  & 15) - 8) * s);
      }
    }
  };

  v8f acc[2][4] = {};
  issueA(0, 0);
  loadW(0, 0);

  const int NKB = IN_FEATURES / BK;
  for (int kb = 0; kb < NKB; ++kb) {
    const int cur = kb & 1;
    wait_async0();
    __syncthreads();
    if (kb + 1 < NKB) {
      issueA(cur ^ 1, (kb + 1) * BK);
      loadW(cur ^ 1, (kb + 1) * BK);
    }

    v16bf af[2], bfr[4];
#pragma unroll
    for (int i = 0; i < 2; ++i) af[i]  = load_frag(sa[cur], waveM + i * 16, lane);
#pragma unroll
    for (int j = 0; j < 4; ++j) bfr[j] = load_frag(sw[cur], waveN + j * 16, lane);
#pragma unroll
    for (int i = 0; i < 2; ++i)
#pragma unroll
      for (int j = 0; j < 4; ++j)
        acc[i][j] = __builtin_amdgcn_wmma_f32_16x16x32_bf16(
            false, af[i], false, bfr[j], (short)0, acc[i][j], false, false);
  }

  const int ln = lane & 15;
  const int hi = lane >> 4;
#pragma unroll
  for (int i = 0; i < 2; ++i)
#pragma unroll
    for (int j = 0; j < 4; ++j) {
      const int n  = colBase + waveN + j * 16 + ln;
      const float bv = bias[n];
#pragma unroll
      for (int g = 0; g < 8; ++g) {
        const int m = rowBase + waveM + i * 16 + hi * 8 + g;
        out[(size_t)m * OUT_FEATURES + n] = acc[i][j][g] + bv;
      }
    }
}

// ===========================================================================
// Fallback kernels (small workspace): inline fp32->bf16 conversion, no async.
// ===========================================================================
__global__ __launch_bounds__(256)
void hadamard_gemm_fb(const float* __restrict__ x,
                      const float* __restrict__ h,
                      __bf16* __restrict__ xr) {
  __shared__ __align__(16) __bf16 lds_a[BM * LDK];
  __shared__ __align__(16) __bf16 lds_b[BN * LDK];

  const int t     = threadIdx.x;
  const int lane  = t & 31;
  const int wid   = t >> 5;
  const int waveM = (wid & 3) * 32;
  const int waveN = (wid >> 2) * 64;
  const int rowBase = blockIdx.y * BM;
  const int colBase = blockIdx.x * BN;

  v8f acc[2][4] = {};

  for (int k0 = 0; k0 < IN_FEATURES; k0 += BK) {
    __syncthreads();
#pragma unroll
    for (int i = 0; i < 4; ++i) {
      const int idx = t + i * 256;
      const int row = idx >> 3;
      const int c4  = idx & 7;
      const float4 v = *(const float4*)(x + (size_t)(rowBase + row) * IN_FEATURES + k0 + c4 * 4);
      __bf16* d = lds_a + row * LDK + c4 * 4;
      d[0] = (__bf16)v.x; d[1] = (__bf16)v.y; d[2] = (__bf16)v.z; d[3] = (__bf16)v.w;
    }
#pragma unroll
    for (int i = 0; i < 4; ++i) {
      const int idx = t + i * 256;
      const int kr  = idx >> 5;
      const int c4  = idx & 31;
      const float4 v = *(const float4*)(h + (size_t)(k0 + kr) * OUT_FEATURES + colBase + c4 * 4);
      const int n = c4 * 4;
      lds_b[(n + 0) * LDK + kr] = (__bf16)v.x;
      lds_b[(n + 1) * LDK + kr] = (__bf16)v.y;
      lds_b[(n + 2) * LDK + kr] = (__bf16)v.z;
      lds_b[(n + 3) * LDK + kr] = (__bf16)v.w;
    }
    __syncthreads();

    v16bf af[2], bfr[4];
#pragma unroll
    for (int i = 0; i < 2; ++i) af[i]  = load_frag(lds_a, waveM + i * 16, lane);
#pragma unroll
    for (int j = 0; j < 4; ++j) bfr[j] = load_frag(lds_b, waveN + j * 16, lane);
#pragma unroll
    for (int i = 0; i < 2; ++i)
#pragma unroll
      for (int j = 0; j < 4; ++j)
        acc[i][j] = __builtin_amdgcn_wmma_f32_16x16x32_bf16(
            false, af[i], false, bfr[j], (short)0, acc[i][j], false, false);
  }

  const int ln = lane & 15;
  const int hi = lane >> 4;
#pragma unroll
  for (int i = 0; i < 2; ++i)
#pragma unroll
    for (int j = 0; j < 4; ++j)
#pragma unroll
      for (int g = 0; g < 8; ++g) {
        const int m = rowBase + waveM + i * 16 + hi * 8 + g;
        const int n = colBase + waveN + j * 16 + ln;
        xr[(size_t)m * OUT_FEATURES + n] = (__bf16)acc[i][j][g];
      }
}

__global__ __launch_bounds__(256)
void qlinear_gemm_fb(const __bf16* __restrict__ xr,
                     const int*    __restrict__ wq,
                     const float*  __restrict__ scales,
                     const float*  __restrict__ bias,
                     float* __restrict__ out) {
  __shared__ __align__(16) __bf16 lds_a[BM * LDK];
  __shared__ __align__(16) __bf16 lds_b[BN * LDK];

  const int t     = threadIdx.x;
  const int lane  = t & 31;
  const int wid   = t >> 5;
  const int waveM = (wid & 3) * 32;
  const int waveN = (wid >> 2) * 64;
  const int rowBase = blockIdx.y * BM;
  const int colBase = blockIdx.x * BN;

  v8f acc[2][4] = {};

  for (int k0 = 0; k0 < IN_FEATURES; k0 += BK) {
    __syncthreads();
#pragma unroll
    for (int i = 0; i < 2; ++i) {
      const int idx = t + i * 256;
      const int row = idx >> 2;
      const int c   = idx & 3;
      const v8bf v = *(const v8bf*)(xr + (size_t)(rowBase + row) * IN_FEATURES + k0 + c * 8);
      *(v8bf*)(lds_a + row * LDK + c * 8) = v;
    }
#pragma unroll
    for (int i = 0; i < 2; ++i) {
      const int idx = t + i * 256;
      const int o   = idx >> 2;
      const int c   = idx & 3;
      const int og  = colBase + o;
      const int4 b4 = *(const int4*)(wq + (size_t)og * (IN_FEATURES / 2) + (k0 >> 1) + c * 4);
      const float s = scales[og * (IN_FEATURES / GROUP_SIZE) + (k0 >> 7)];
      __bf16* d = lds_b + o * LDK + c * 8;
      const int bb[4] = { b4.x, b4.y, b4.z, b4.w };
#pragma unroll
      for (int q = 0; q < 4; ++q) {
        const int b = bb[q];
        d[2 * q + 0] = (__bf16)((float)(( b        & 15) - 8) * s);
        d[2 * q + 1] = (__bf16)((float)(((b >> 4)  & 15) - 8) * s);
      }
    }
    __syncthreads();

    v16bf af[2], bfr[4];
#pragma unroll
    for (int i = 0; i < 2; ++i) af[i]  = load_frag(lds_a, waveM + i * 16, lane);
#pragma unroll
    for (int j = 0; j < 4; ++j) bfr[j] = load_frag(lds_b, waveN + j * 16, lane);
#pragma unroll
    for (int i = 0; i < 2; ++i)
#pragma unroll
      for (int j = 0; j < 4; ++j)
        acc[i][j] = __builtin_amdgcn_wmma_f32_16x16x32_bf16(
            false, af[i], false, bfr[j], (short)0, acc[i][j], false, false);
  }

  const int ln = lane & 15;
  const int hi = lane >> 4;
#pragma unroll
  for (int i = 0; i < 2; ++i)
#pragma unroll
    for (int j = 0; j < 4; ++j) {
      const int n  = colBase + waveN + j * 16 + ln;
      const float bv = bias[n];
#pragma unroll
      for (int g = 0; g < 8; ++g) {
        const int m = rowBase + waveM + i * 16 + hi * 8 + g;
        out[(size_t)m * OUT_FEATURES + n] = acc[i][j][g] + bv;
      }
    }
}

// ---------------------------------------------------------------------------
extern "C" void kernel_launch(void* const* d_in, const int* in_sizes, int n_in,
                              void* d_out, int out_size, void* d_ws, size_t ws_size,
                              hipStream_t stream) {
  (void)in_sizes; (void)n_in; (void)out_size;
  const float* x  = (const float*)d_in[0];  // [2,2048,4096] fp32
  const float* h  = (const float*)d_in[1];  // [4096,4096]   fp32
  const float* sc = (const float*)d_in[2];  // [4096,32]     fp32
  const float* bs = (const float*)d_in[3];  // [4096]        fp32
  const int*   wq = (const int*)d_in[4];    // [4096,2048]   int32 (packed nibbles)
  float* out = (float*)d_out;               // [2,2048,4096] fp32

  const size_t N_X  = (size_t)NROWS * IN_FEATURES;
  const size_t N_H  = (size_t)IN_FEATURES * OUT_FEATURES;
  const size_t N_XR = (size_t)NROWS * OUT_FEATURES;
  const size_t need_fast = (N_X + N_H + N_XR) * sizeof(__bf16);  // 96 MB

  dim3 block(256);
  dim3 grid(OUT_FEATURES / BN, NROWS / BM);  // 32 x 32

  if (ws_size >= need_fast) {
    __bf16* xb = (__bf16*)d_ws;
    __bf16* ht = xb + N_X;
    __bf16* xr = ht + N_H;
    cvt_f32_bf16<<<dim3(N_X / (256 * 16)), block, 0, stream>>>(x, xb);
    transpose_cvt_f32_bf16<<<dim3(OUT_FEATURES / TT, IN_FEATURES / TT), block, 0, stream>>>(h, ht);
    gemm_bf16_async<<<grid, block, 0, stream>>>(xb, ht, xr);
    qlinear_gemm_async<<<grid, block, 0, stream>>>(xr, wq, sc, bs, out);
  } else {
    __bf16* xr = (__bf16*)d_ws;  // needs 32 MB
    hadamard_gemm_fb<<<grid, block, 0, stream>>>(x, h, xr);
    qlinear_gemm_fb<<<grid, block, 0, stream>>>(xr, wq, sc, bs, out);
  }
}
